// MyGATv2EncoderVAEOLD_62448824484499
// MI455X (gfx1250) — compile-verified, matrix operand order checked by reading, not run on hardware
//
#include <hip/hip_runtime.h>
#include <hip/hip_bf16.h>
#include <math.h>

// ---------------------------------------------------------------------------
// GATv2 encoder VAE forward, gfx1250.
// Dense transforms use v_wmma_f32_16x16x32_f16 (wave32 WMMA) with weight
// panels pre-swizzled into the CDNA5 per-lane B-fragment layout so operand
// loads are 128-bit vector loads. Edge attention uses L2 float atomics.
// ---------------------------------------------------------------------------

#define N_NODES 100000
#define N_EDGES 1600000
#define ETOT    (N_EDGES + N_NODES)   // edges + self loops
#define LDF     64                    // padded per-node feature stride
#define BN_EPS  1e-5f

typedef __attribute__((ext_vector_type(16))) _Float16 v16h;
typedef __attribute__((ext_vector_type(8)))  _Float16 v8h;
typedef __attribute__((ext_vector_type(8)))  float    v8f;

// ---------------------------------------------------------------------------
// helpers
// ---------------------------------------------------------------------------
__device__ __forceinline__ void atomicMaxFloat(float* addr, float val) {
    // order-preserving int trick; emax initialized to -3e38 (negative)
    if (val >= 0.0f) atomicMax((int*)addr, __float_as_int(val));
    else             atomicMin((unsigned int*)addr, __float_as_uint(val));
}

// ---------------------------------------------------------------------------
// 0) pack raw input x [N,22] -> hbuf f32 [N,64] (zero padded) + xh f16 [N,64]
// ---------------------------------------------------------------------------
__global__ void k_pack_input(const float* __restrict__ x,
                             float* __restrict__ hbuf,
                             _Float16* __restrict__ xh,
                             int cin) {
    int idx = blockIdx.x * blockDim.x + threadIdx.x;
    if (idx >= N_NODES * LDF) return;
    int n = idx >> 6, j = idx & 63;
    float v = (j < cin) ? x[(size_t)n * cin + j] : 0.0f;
    hbuf[idx] = v;
    xh[idx]   = (_Float16)v;
}

// ---------------------------------------------------------------------------
// 1) weight prep: swizzle Wl/Wr [cin,hc] f32 into the exact per-lane WMMA
//    B-fragment layout, f16, K padded to 64, cols 0..hc-1 = Wl,
//    cols 64..64+hc-1 = Wr (zero elsewhere).
//    wfrag[kstep][tile][lane][elem]:
//      klo  = kstep*32 + (lane>=16 ? 8 : 0)
//      K    = elem<8 ? klo+elem : klo+16+(elem-8)
//      col  = tile*16 + (lane&15)
//    Also builds biasbuf[128]: bl in cols 0..hc, br in cols 64..64+hc, else 0.
// ---------------------------------------------------------------------------
__global__ void k_prep_frag(const float* __restrict__ Wl,
                            const float* __restrict__ Wr,
                            const float* __restrict__ bl,
                            const float* __restrict__ br,
                            _Float16* __restrict__ wfrag,
                            float* __restrict__ biasbuf,
                            int cin, int hc) {
    int idx = blockIdx.x * blockDim.x + threadIdx.x;
    if (idx < 128) {
        float bv = 0.0f;
        if (idx < hc)                          bv = bl[idx];
        else if (idx >= 64 && idx - 64 < hc)   bv = br[idx - 64];
        biasbuf[idx] = bv;
    }
    if (idx >= 2 * 8 * 32 * 16) return;
    int elem    = idx & 15;
    int lane    = (idx >> 4) & 31;
    int tile    = (idx >> 9) & 7;
    int kstep   = idx >> 12;
    int halfsel = lane >> 4;
    int lm      = lane & 15;
    int klo     = kstep * 32 + halfsel * 8;
    int k       = (elem < 8) ? (klo + elem) : (klo + 16 + (elem - 8));
    int col     = tile * 16 + lm;
    float v = 0.0f;
    if (k < cin) {
        if (col < hc)                          v = Wl[(size_t)k * hc + col];
        else if (col >= 64 && col - 64 < hc)   v = Wr[(size_t)k * hc + (col - 64)];
    }
    wfrag[idx] = (_Float16)v;
}

// ---------------------------------------------------------------------------
// 2) WMMA GEMM: xh [N,64]f16 x W [64,128]f16 -> xl [N,64]f32, xr [N,64]f32
//    (bias folded in). block = 256 threads = 8 waves; wave w -> col tile
//    [16w,16w+16); grid = N/16 row tiles. All operand loads are b128.
// ---------------------------------------------------------------------------
__global__ void k_gemm_wmma(const _Float16* __restrict__ xh,
                            const _Float16* __restrict__ wfrag,
                            const float* __restrict__ biasbuf,
                            float* __restrict__ xl,
                            float* __restrict__ xr) {
    const int wave    = threadIdx.x >> 5;
    const int lane    = threadIdx.x & 31;
    const int halfsel = lane >> 4;        // 0: lanes 0-15, 1: lanes 16-31
    const int lm      = lane & 15;
    const int m0      = blockIdx.x * 16;  // N divisible by 16

    const _Float16* arow = xh + (size_t)(m0 + lm) * LDF;
    v8f acc = {};

#pragma unroll
    for (int kstep = 0; kstep < 2; ++kstep) {
        const int klo = kstep * 32 + halfsel * 8;
        // A fragment: two contiguous 16B chunks of this lane's row
        v8h alo = *(const v8h*)(arow + klo);
        v8h ahi = *(const v8h*)(arow + klo + 16);
        // B fragment: pre-swizzled, 16 contiguous halves per lane
        const _Float16* bf = wfrag + (((size_t)kstep * 8 + wave) * 32 + lane) * 16;
        v8h blo = *(const v8h*)(bf);
        v8h bhi = *(const v8h*)(bf + 8);
        v16h a, b;
#pragma unroll
        for (int j = 0; j < 8; ++j) {
            a[j] = alo[j]; a[j + 8] = ahi[j];
            b[j] = blo[j]; b[j + 8] = bhi[j];
        }
        acc = __builtin_amdgcn_wmma_f32_16x16x32_f16(
            /*neg_a=*/false, a, /*neg_b=*/false, b,
            /*c_mod=*/(short)0, acc, /*reuse_a=*/false, /*reuse_b=*/false);
    }

    // D layout: VGPR r holds row m0 + r + 8*halfsel, col wave*16 + lm
    const int g      = wave * 16 + lm;
    const float bias = biasbuf[g];
    const int rbase  = m0 + halfsel * 8;
    float* outp = (wave < 4) ? (xl + (size_t)rbase * LDF + g)
                             : (xr + (size_t)rbase * LDF + (g - 64));
#pragma unroll
    for (int r = 0; r < 8; ++r)
        outp[(size_t)r * LDF] = acc[r] + bias;
}

// ---------------------------------------------------------------------------
// 3) per-layer init: agg = 0, emax = -3e38, den = 0
// ---------------------------------------------------------------------------
__global__ void k_init_layer(float* __restrict__ emax,
                             float* __restrict__ den,
                             float* __restrict__ agg) {
    int idx = blockIdx.x * blockDim.x + threadIdx.x;
    if (idx >= N_NODES * LDF) return;
    agg[idx] = 0.0f;
    int j = idx & 63;
    if (j < 8) {
        int n = idx >> 6;
        emax[n * 8 + j] = -3e38f;
        den[n * 8 + j]  = 0.0f;
    }
}

// ---------------------------------------------------------------------------
// 4) edge pass 1: e[eid,h] = sum_c att[h,c] * lrelu(xl[src]+xr[dst]); seg-max
// ---------------------------------------------------------------------------
__global__ void k_edge_e(const int* __restrict__ ei,
                         const float* __restrict__ xl,
                         const float* __restrict__ xr,
                         const float* __restrict__ att,
                         float* __restrict__ ews,
                         float* __restrict__ emax,
                         int H, int C) {
    int e = blockIdx.x * blockDim.x + threadIdx.x;
    if (e >= ETOT) return;
    // stream-prefetch the sequential edge-index reads a few blocks ahead
    __builtin_prefetch(ei + e + 8 * 256, 0, 1);
    __builtin_prefetch(ei + N_EDGES + e + 8 * 256, 0, 1);
    int src, dst;
    if (e < N_EDGES) { src = ei[e]; dst = ei[N_EDGES + e]; }
    else             { src = dst = e - N_EDGES; }
    const float* pl = xl + (size_t)src * LDF;
    const float* pr = xr + (size_t)dst * LDF;
    for (int h = 0; h < H; ++h) {
        float s = 0.0f;
        for (int c = 0; c < C; ++c) {
            float v = pl[h * C + c] + pr[h * C + c];
            v = (v > 0.0f) ? v : 0.2f * v;   // leaky_relu 0.2
            s += att[h * C + c] * v;
        }
        ews[(size_t)e * H + h] = s;
        atomicMaxFloat(&emax[dst * 8 + h], s);
    }
}

// ---------------------------------------------------------------------------
// 5) edge pass 2: ex = exp(e - emax[dst]); seg-sum into den
// ---------------------------------------------------------------------------
__global__ void k_edge_exp(const int* __restrict__ ei,
                           float* __restrict__ ews,
                           const float* __restrict__ emax,
                           float* __restrict__ den,
                           int H) {
    int e = blockIdx.x * blockDim.x + threadIdx.x;
    if (e >= ETOT) return;
    int dst = (e < N_EDGES) ? ei[N_EDGES + e] : (e - N_EDGES);
    for (int h = 0; h < H; ++h) {
        float ex = __expf(ews[(size_t)e * H + h] - emax[dst * 8 + h]);
        ews[(size_t)e * H + h] = ex;
        atomicAdd(&den[dst * 8 + h], ex);
    }
}

// ---------------------------------------------------------------------------
// 6) edge pass 3: agg[dst] += xl[src] * (ex / den[dst])
// ---------------------------------------------------------------------------
__global__ void k_edge_agg(const int* __restrict__ ei,
                           const float* __restrict__ xl,
                           const float* __restrict__ ews,
                           const float* __restrict__ den,
                           float* __restrict__ agg,
                           int H, int C) {
    int e = blockIdx.x * blockDim.x + threadIdx.x;
    if (e >= ETOT) return;
    int src, dst;
    if (e < N_EDGES) { src = ei[e]; dst = ei[N_EDGES + e]; }
    else             { src = dst = e - N_EDGES; }
    const float* pl = xl + (size_t)src * LDF;
    float* pa = agg + (size_t)dst * LDF;
    for (int h = 0; h < H; ++h) {
        float alpha = ews[(size_t)e * H + h] / den[dst * 8 + h];
        for (int c = 0; c < C; ++c)
            atomicAdd(&pa[h * C + c], pl[h * C + c] * alpha);
    }
}

// ---------------------------------------------------------------------------
// 7) post: concat/mean + bias + batchnorm (+ optional ELU); writes next
//    features (f32 padded) and f16 copy for the next WMMA GEMM.
// ---------------------------------------------------------------------------
__global__ void k_post(const float* __restrict__ agg,
                       const float* __restrict__ b,
                       const float* __restrict__ gam,
                       const float* __restrict__ bet,
                       const float* __restrict__ rm,
                       const float* __restrict__ rv,
                       float* __restrict__ hbuf,
                       _Float16* __restrict__ xh,
                       int H, int C, int concat, int do_elu) {
    int idx = blockIdx.x * blockDim.x + threadIdx.x;
    if (idx >= N_NODES * LDF) return;
    int n = idx >> 6, j = idx & 63;
    int outd = concat ? H * C : C;
    float y = 0.0f;
    if (j < outd) {
        if (concat) {
            y = agg[(size_t)n * LDF + j];
        } else {
            float s = 0.0f;
            for (int h = 0; h < H; ++h) s += agg[(size_t)n * LDF + h * C + j];
            y = s / (float)H;
        }
        y += b[j];
        y = gam[j] * (y - rm[j]) * rsqrtf(rv[j] + BN_EPS) + bet[j];
        if (do_elu) y = (y > 0.0f) ? y : (__expf(y) - 1.0f);
    }
    hbuf[idx] = y;
    xh[idx]   = (_Float16)y;
}

// ---------------------------------------------------------------------------
// 8) final 10x10 heads: mu = h@Wmu + bmu ; lv = h@Wlv + blv
//    out layout: [N,10] mu then [N,10] log_var, concatenated flat.
// ---------------------------------------------------------------------------
__global__ void k_final(const float* __restrict__ hbuf,
                        const float* __restrict__ Wmu,
                        const float* __restrict__ bmu,
                        const float* __restrict__ Wlv,
                        const float* __restrict__ blv,
                        float* __restrict__ out) {
    int n = blockIdx.x * blockDim.x + threadIdx.x;
    if (n >= N_NODES) return;
    float h[10];
#pragma unroll
    for (int k = 0; k < 10; ++k) h[k] = hbuf[(size_t)n * LDF + k];
#pragma unroll
    for (int j = 0; j < 10; ++j) {
        float mu = bmu[j], lv = blv[j];
#pragma unroll
        for (int k = 0; k < 10; ++k) {
            mu += h[k] * Wmu[k * 10 + j];
            lv += h[k] * Wlv[k * 10 + j];
        }
        out[(size_t)n * 10 + j]                        = mu;
        out[(size_t)N_NODES * 10 + (size_t)n * 10 + j] = lv;
    }
}

// ---------------------------------------------------------------------------
// host launcher
// ---------------------------------------------------------------------------
extern "C" void kernel_launch(void* const* d_in, const int* in_sizes, int n_in,
                              void* d_out, int out_size, void* d_ws, size_t ws_size,
                              hipStream_t stream) {
    (void)in_sizes; (void)n_in; (void)out_size; (void)ws_size;

    const float* x   = (const float*)d_in[0];
    const int*   ei  = (const int*)d_in[1];
    const float* Wmu = (const float*)d_in[42];
    const float* bmu = (const float*)d_in[43];
    const float* Wlv = (const float*)d_in[44];
    const float* blv = (const float*)d_in[45];

    // workspace carve-out
    char* ws = (char*)d_ws;
    size_t off = 0;
    auto carve = [&](size_t bytes) { char* p = ws + off; off += (bytes + 255) & ~(size_t)255; return p; };
    _Float16* xh      = (_Float16*)carve((size_t)N_NODES * LDF * sizeof(_Float16));
    float*    hbuf    = (float*)   carve((size_t)N_NODES * LDF * sizeof(float));
    float*    xl      = (float*)   carve((size_t)N_NODES * LDF * sizeof(float));
    float*    xr      = (float*)   carve((size_t)N_NODES * LDF * sizeof(float));
    float*    agg     = (float*)   carve((size_t)N_NODES * LDF * sizeof(float));
    float*    ews     = (float*)   carve((size_t)ETOT * 5 * sizeof(float));
    float*    emax    = (float*)   carve((size_t)N_NODES * 8 * sizeof(float));
    float*    den     = (float*)   carve((size_t)N_NODES * 8 * sizeof(float));
    _Float16* wfrag   = (_Float16*)carve((size_t)2 * 8 * 32 * 16 * sizeof(_Float16));
    float*    biasbuf = (float*)   carve((size_t)128 * sizeof(float));

    const int TB  = 256;
    const int gNF = (N_NODES * LDF + TB - 1) / TB;   // node*feature grids
    const int gE  = (ETOT + TB - 1) / TB;            // edge grids
    const int gM  = N_NODES / 16;                    // GEMM row tiles (exact)
    const int gN  = (N_NODES + TB - 1) / TB;

    // pack input features
    k_pack_input<<<gNF, TB, 0, stream>>>(x, hbuf, xh, 22);

    struct LP { int cin, H, C, concat; };
    const LP layers[4] = { {22, 3, 18, 1}, {54, 3, 14, 1}, {42, 3, 12, 1}, {36, 5, 10, 0} };

    for (int i = 0; i < 4; ++i) {
        const int base = 2 + 10 * i;
        const float* Wl  = (const float*)d_in[base + 0];
        const float* Wr  = (const float*)d_in[base + 1];
        const float* bl  = (const float*)d_in[base + 2];
        const float* br  = (const float*)d_in[base + 3];
        const float* att = (const float*)d_in[base + 4];
        const float* b   = (const float*)d_in[base + 5];
        const float* g   = (const float*)d_in[base + 6];
        const float* be  = (const float*)d_in[base + 7];
        const float* rm  = (const float*)d_in[base + 8];
        const float* rv  = (const float*)d_in[base + 9];
        const int H = layers[i].H, C = layers[i].C, hc = H * C;

        k_prep_frag<<<(2 * 8 * 32 * 16 + TB - 1) / TB, TB, 0, stream>>>(
            Wl, Wr, bl, br, wfrag, biasbuf, layers[i].cin, hc);
        k_gemm_wmma<<<gM, TB, 0, stream>>>(xh, wfrag, biasbuf, xl, xr);
        k_init_layer<<<gNF, TB, 0, stream>>>(emax, den, agg);
        k_edge_e<<<gE, TB, 0, stream>>>(ei, xl, xr, att, ews, emax, H, C);
        k_edge_exp<<<gE, TB, 0, stream>>>(ei, ews, emax, den, H);
        k_edge_agg<<<gE, TB, 0, stream>>>(ei, xl, ews, den, agg, H, C);
        k_post<<<gNF, TB, 0, stream>>>(agg, b, g, be, rm, rv, hbuf, xh,
                                       H, C, layers[i].concat, /*elu only layer 1*/ i == 0);
    }

    k_final<<<gN, TB, 0, stream>>>(hbuf, Wmu, bmu, Wlv, blv, (float*)d_out);
}
